// Decoder_21234318311991
// MI455X (gfx1250) — compile-verified, matrix operand order checked by reading, not run on hardware
//
#include <hip/hip_runtime.h>
#include <math.h>

// ---------------------------------------------------------------------------
// Pointer-network decoder, MI455X (gfx1250, wave32, WMMA f16 16x16x32).
// B=256 batch, S=512 steps, H=512 hidden, A=512 actions.
// Weights packed once into WMMA per-lane B-tile layout (ISA 7.12.2); steady
// state per step = 3 fused kernels driven by v_wmma_f32_16x16x32_f16.
// ---------------------------------------------------------------------------

typedef _Float16 f16;
typedef __attribute__((ext_vector_type(8)))  f16   v8h;
typedef __attribute__((ext_vector_type(16))) f16   v16h;
typedef __attribute__((ext_vector_type(8)))  float v8f;

#define BQ 256
#define SQ 512
#define HQ 512
#define AQ 512

__device__ __forceinline__ float sigmoidf_(float x) { return 1.0f / (1.0f + __expf(-x)); }

// ---------------------------------------------------------------------------
// One-time: initial state. c = c0 (f32), h = h0 (f16), mask = 0, dec_in = 1.
// ---------------------------------------------------------------------------
__global__ void init_state(const float* __restrict__ h0, const float* __restrict__ c0,
                           f16* __restrict__ hbuf0, float* __restrict__ cbuf,
                           float* __restrict__ dec_in, int* __restrict__ prev,
                           unsigned char* __restrict__ mask)
{
    int idx = blockIdx.x * blockDim.x + threadIdx.x;
    if (idx < BQ * HQ) {
        cbuf[idx]  = c0[idx];
        hbuf0[idx] = (f16)h0[idx];
        mask[idx]  = 0;               // B*A == B*H
    }
    if (idx < BQ) { dec_in[idx] = 1.0f; prev[idx] = 0; }
}

// ---------------------------------------------------------------------------
// One-time: pack combined gate weights into WMMA B-tile layout.
// Logical B[k,n] (K=2H x N=4H): k<H -> W_ih[n,k]; k>=H -> W_hh[n,k-H].
// Tile (kt,nt) of 32x16; lane l holds col n=nt*16+(l&15), 16 f16 contiguous:
//   K = kt*32 + koff+{0..7}, then kt*32 + koff+16+{0..7}, koff = (l&16)?8:0
// Storage index == ((kt*128+nt)*32 + l)*16 + j  (32B-aligned per lane).
// ---------------------------------------------------------------------------
__global__ void pack_wg(const float* __restrict__ Wih, const float* __restrict__ Whh,
                        f16* __restrict__ WgP)
{
    int idx = blockIdx.x * blockDim.x + threadIdx.x;     // 2^21 elements
    if (idx >= 32 * 128 * 32 * 16) return;
    int j  = idx & 15;
    int l  = (idx >> 4)  & 31;
    int nt = (idx >> 9)  & 127;
    int kt =  idx >> 16;
    int n    = nt * 16 + (l & 15);
    int koff = (l & 16) ? 8 : 0;
    int kl   = (j < 8) ? (koff + j) : (koff + 16 + (j - 8));
    int k    = kt * 32 + kl;
    float v  = (k < HQ) ? Wih[n * HQ + k] : Whh[n * HQ + (k - HQ)];
    WgP[idx] = (f16)v;
}

// Same packing for out_W (K=H x N=A): B[k,n] = out_W[n,k].
__global__ void pack_wout(const float* __restrict__ outW, f16* __restrict__ WoP)
{
    int idx = blockIdx.x * blockDim.x + threadIdx.x;     // 2^18 elements
    if (idx >= 16 * 32 * 32 * 16) return;
    int j  = idx & 15;
    int l  = (idx >> 4) & 31;
    int nt = (idx >> 9) & 31;
    int kt =  idx >> 14;
    int n    = nt * 16 + (l & 15);
    int koff = (l & 16) ? 8 : 0;
    int kl   = (j < 8) ? (koff + j) : (koff + 16 + (j - 8));
    int k    = kt * 32 + kl;
    WoP[idx] = (f16)outW[n * HQ + k];
}

// ---------------------------------------------------------------------------
// Per step: x = relu(dec_in * emb_W + emb_b) -> f16   [B,H]
// ---------------------------------------------------------------------------
__global__ void prep_x(const float* __restrict__ dec_in, const float* __restrict__ embW,
                       const float* __restrict__ embB, f16* __restrict__ xbuf)
{
    int idx = blockIdx.x * blockDim.x + threadIdx.x;
    if (idx >= BQ * HQ) return;
    int b = idx >> 9;
    int h = idx & (HQ - 1);
    float x = dec_in[b] * embW[h] + embB[h];
    xbuf[idx] = (f16)(x > 0.0f ? x : 0.0f);
}

// ---------------------------------------------------------------------------
// Per step: gates GEMM [B,2H]x[2H,4H] via WMMA + fused LSTM cell epilogue.
// Grid (16,8) x 128 threads (4 waves). Wave handles 16 batch rows x 16 h-cols
// across all 4 gates (i,f,g,o) -> 4 accumulators, K-loop of 32 WMMA steps.
// h double-buffered (hin/hout) to avoid cross-block RAW on h.
// ---------------------------------------------------------------------------
__global__ void __launch_bounds__(128)
gates_cell(const f16* __restrict__ xbuf, const f16* __restrict__ hin,
           const f16* __restrict__ WgP,
           const float* __restrict__ bih, const float* __restrict__ bhh,
           float* __restrict__ cbuf, f16* __restrict__ hout)
{
    const int lane = threadIdx.x & 31;
    const int wave = threadIdx.x >> 5;
    const int btile = blockIdx.x;                 // 16 batch rows
    const int n0    = blockIdx.y * 64 + wave * 16;// h-column base
    const int m     = lane & 15;
    const int koff  = (lane & 16) ? 8 : 0;
    const int row   = btile * 16 + m;
    const int ntBase = n0 >> 4;                   // 0..31 within one gate block

    v8f accI = {}, accF = {}, accG = {}, accO = {};

    for (int kt = 0; kt < 32; ++kt) {
        // A tile: 16x32 f16, K<512 from x, K>=512 from h_in (tiles never straddle).
        const f16* arow = (kt < 16) ? (xbuf + (size_t)row * HQ + kt * 32)
                                    : (hin  + (size_t)row * HQ + (kt - 16) * 32);
        v8h alo = *(const v8h*)(arow + koff);
        v8h ahi = *(const v8h*)(arow + koff + 16);
        v16h a  = __builtin_shufflevector(alo, ahi, 0,1,2,3,4,5,6,7,8,9,10,11,12,13,14,15);

        const size_t kb = (size_t)kt * 128;       // tile row base (128 N-tiles)
        v16h bvI = *(const v16h*)(WgP + (kb + (size_t)( 0 + ntBase)) * 512 + lane * 16);
        v16h bvF = *(const v16h*)(WgP + (kb + (size_t)(32 + ntBase)) * 512 + lane * 16);
        v16h bvG = *(const v16h*)(WgP + (kb + (size_t)(64 + ntBase)) * 512 + lane * 16);
        v16h bvO = *(const v16h*)(WgP + (kb + (size_t)(96 + ntBase)) * 512 + lane * 16);
        if (kt + 1 < 32)  // CDNA5 global_prefetch_b8 for next K-tile
            __builtin_prefetch(WgP + ((size_t)(kt + 1) * 128 + ntBase) * 512 + lane * 16, 0, 1);

        accI = __builtin_amdgcn_wmma_f32_16x16x32_f16(false, a, false, bvI, (short)0, accI, false, false);
        accF = __builtin_amdgcn_wmma_f32_16x16x32_f16(false, a, false, bvF, (short)0, accF, false, false);
        accG = __builtin_amdgcn_wmma_f32_16x16x32_f16(false, a, false, bvG, (short)0, accG, false, false);
        accO = __builtin_amdgcn_wmma_f32_16x16x32_f16(false, a, false, bvO, (short)0, accO, false, false);
    }

    // Epilogue: C/D layout -> lane col = lane&15 (same h for all 8 elems),
    // row = (lane&16 ? 8 : 0) + e. Torch gate order i,f,g,o.
    const int h  = n0 + m;
    const float bI = bih[0 * HQ + h] + bhh[0 * HQ + h];
    const float bF = bih[1 * HQ + h] + bhh[1 * HQ + h];
    const float bG = bih[2 * HQ + h] + bhh[2 * HQ + h];
    const float bO = bih[3 * HQ + h] + bhh[3 * HQ + h];
    const int rbase = (lane & 16) ? 8 : 0;
#pragma unroll
    for (int e = 0; e < 8; ++e) {
        int brow = btile * 16 + rbase + e;
        float ig = sigmoidf_(accI[e] + bI);
        float fg = sigmoidf_(accF[e] + bF);
        float gg = tanhf   (accG[e] + bG);
        float og = sigmoidf_(accO[e] + bO);
        float cn = fg * cbuf[(size_t)brow * HQ + h] + ig * gg;
        cbuf[(size_t)brow * HQ + h] = cn;
        hout[(size_t)brow * HQ + h] = (f16)(og * tanhf(cn));
    }
}

// ---------------------------------------------------------------------------
// Per step: logits GEMM [B,H]x[H,A] via WMMA into LDS, then fused per-row
// raw-argmax (-> next dec_in), mask, masked-argmax (-> chosen), log-softmax.
// Grid 16 blocks (16 rows each) x 256 threads (8 waves, 4 N-tiles per wave).
// ---------------------------------------------------------------------------
__global__ void __launch_bounds__(256)
logits_select(const f16* __restrict__ hcur, const f16* __restrict__ WoP,
              const float* __restrict__ outb,
              float* __restrict__ out, float* __restrict__ dec_in,
              int* __restrict__ prev, unsigned char* __restrict__ mask, int t)
{
    __shared__ float lds[16][AQ];   // 32 KB
    const int lane  = threadIdx.x & 31;
    const int wave  = threadIdx.x >> 5;
    const int btile = blockIdx.x;
    const int m     = lane & 15;
    const int koff  = (lane & 16) ? 8 : 0;
    const int row   = btile * 16 + m;

    v8f acc[4] = {{}, {}, {}, {}};
    for (int kt = 0; kt < 16; ++kt) {
        const f16* arow = hcur + (size_t)row * HQ + kt * 32;
        v8h alo = *(const v8h*)(arow + koff);
        v8h ahi = *(const v8h*)(arow + koff + 16);
        v16h a  = __builtin_shufflevector(alo, ahi, 0,1,2,3,4,5,6,7,8,9,10,11,12,13,14,15);
#pragma unroll
        for (int j = 0; j < 4; ++j) {
            int nt = wave * 4 + j;
            v16h bv = *(const v16h*)(WoP + ((size_t)kt * 32 + nt) * 512 + lane * 16);
            acc[j] = __builtin_amdgcn_wmma_f32_16x16x32_f16(false, a, false, bv, (short)0, acc[j], false, false);
        }
    }
    const int rbase = (lane & 16) ? 8 : 0;
#pragma unroll
    for (int j = 0; j < 4; ++j) {
        int col = (wave * 4 + j) * 16 + m;
        float ob = outb[col];
#pragma unroll
        for (int e = 0; e < 8; ++e) lds[rbase + e][col] = acc[j][e] + ob;
    }
    __syncthreads();

    // Per-row phase: wave w owns rows 2w, 2w+1; lane covers cols lane+32*jj.
    for (int rr = wave * 2; rr < wave * 2 + 2; ++rr) {
        int b = btile * 16 + rr;
        int prevb = prev[b];
        float rawV = -INFINITY; int rawI = 0;
        float mskV = -INFINITY; int mskI = 0;
        float mv[16];
#pragma unroll
        for (int jj = 0; jj < 16; ++jj) {
            int c = lane + 32 * jj;
            float v = lds[rr][c];
            if (v > rawV) { rawV = v; rawI = c; }
            float x;
            if (t == 0) {
                x = (c == 0) ? 1.0f : -INFINITY;      // step-0 override
            } else {
                bool mm = (mask[(size_t)b * AQ + c] != 0) || (c == prevb);
                x = mm ? -INFINITY : v;
            }
            mv[jj] = x;
            if (x > mskV) { mskV = x; mskI = c; }
        }
        // wave32 reductions; ties -> lowest index (matches jnp.argmax).
#pragma unroll
        for (int off = 16; off >= 1; off >>= 1) {
            float ov = __shfl_xor(rawV, off); int oi = __shfl_xor(rawI, off);
            if (ov > rawV || (ov == rawV && oi < rawI)) { rawV = ov; rawI = oi; }
            float pv = __shfl_xor(mskV, off); int pi = __shfl_xor(mskI, off);
            if (pv > mskV || (pv == mskV && pi < mskI)) { mskV = pv; mskI = pi; }
        }
        float s = 0.0f;
#pragma unroll
        for (int jj = 0; jj < 16; ++jj) s += __expf(mv[jj] - mskV);
#pragma unroll
        for (int off = 16; off >= 1; off >>= 1) s += __shfl_xor(s, off);

        if (lane == 0) {
            out[(size_t)b * SQ + t] = (float)mskI;                         // paths [B,1,S]
            out[(size_t)BQ * SQ + (size_t)b * SQ + t] = -__logf(s);        // logp  [B,S,1]
            dec_in[b] = (float)rawI;   // next input = argmax of RAW logits
            prev[b]   = mskI;
            if (t > 0) mask[(size_t)b * AQ + prevb] = 1;  // persist for later steps
        }
    }
}

// ---------------------------------------------------------------------------
extern "C" void kernel_launch(void* const* d_in, const int* in_sizes, int n_in,
                              void* d_out, int out_size, void* d_ws, size_t ws_size,
                              hipStream_t stream)
{
    (void)in_sizes; (void)n_in; (void)out_size; (void)ws_size;
    // setup_inputs order: encoder_output(unused), h0, c0, emb_W, emb_b,
    //                     W_ih, W_hh, b_ih, b_hh, out_W, out_b
    const float* h0   = (const float*)d_in[1];
    const float* c0   = (const float*)d_in[2];
    const float* embW = (const float*)d_in[3];
    const float* embB = (const float*)d_in[4];
    const float* Wih  = (const float*)d_in[5];
    const float* Whh  = (const float*)d_in[6];
    const float* bih  = (const float*)d_in[7];
    const float* bhh  = (const float*)d_in[8];
    const float* outW = (const float*)d_in[9];
    const float* outb = (const float*)d_in[10];
    float* out = (float*)d_out;

    char* ws = (char*)d_ws;
    size_t off = 0;
    auto alloc = [&](size_t bytes) -> char* {
        char* p = ws + off;
        off = (off + bytes + 255) & ~(size_t)255;
        return p;
    };
    f16*   WgP   = (f16*)alloc((size_t)32 * 128 * 32 * 16 * sizeof(f16)); // 4 MiB packed [Wih;Whh]
    f16*   WoP   = (f16*)alloc((size_t)16 * 32  * 32 * 16 * sizeof(f16)); // 512 KiB packed out_W
    f16*   xbuf  = (f16*)alloc((size_t)BQ * HQ * sizeof(f16));
    f16*   hb0   = (f16*)alloc((size_t)BQ * HQ * sizeof(f16));
    f16*   hb1   = (f16*)alloc((size_t)BQ * HQ * sizeof(f16));
    float* cbuf  = (float*)alloc((size_t)BQ * HQ * sizeof(float));
    float* decin = (float*)alloc(BQ * sizeof(float));
    int*   prev  = (int*)alloc(BQ * sizeof(int));
    unsigned char* mask = (unsigned char*)alloc((size_t)BQ * AQ);

    init_state<<<dim3((BQ * HQ + 255) / 256), dim3(256), 0, stream>>>(h0, c0, hb0, cbuf, decin, prev, mask);
    pack_wg  <<<dim3((32 * 128 * 32 * 16) / 256), dim3(256), 0, stream>>>(Wih, Whh, WgP);
    pack_wout<<<dim3((16 * 32  * 32 * 16) / 256), dim3(256), 0, stream>>>(outW, WoP);

    for (int t = 0; t < SQ; ++t) {
        f16* hin  = (t & 1) ? hb1 : hb0;
        f16* hout = (t & 1) ? hb0 : hb1;
        prep_x<<<dim3((BQ * HQ + 255) / 256), dim3(256), 0, stream>>>(decin, embW, embB, xbuf);
        gates_cell<<<dim3(16, 8), dim3(128), 0, stream>>>(xbuf, hin, WgP, bih, bhh, cbuf, hout);
        logits_select<<<dim3(16), dim3(256), 0, stream>>>(hout, WoP, outb, out, decin, prev, mask, t);
    }
}
